// SimpleAdjGNN_69071664054876
// MI455X (gfx1250) — matrix operand order repaired and checked by reading, not compiled, not run on hardware
//
#include <hip/hip_runtime.h>
#include <hip/hip_bf16.h>
#include <math.h>

#define Nn   50000
#define EDIR 1600000
#define EUND 800000
#define Hd   128
#define Ld   3
#define EPSc 1e-6f

typedef __attribute__((ext_vector_type(16))) _Float16 v16h;
typedef __attribute__((ext_vector_type(8)))  _Float16 v8h;
typedef __attribute__((ext_vector_type(8)))  float    v8f;

// D = A(16x32 f16) x B(32x16 f16) + C(16x16 f32)
__device__ __forceinline__ v8f wmma16(v16h a, v16h b, v8f c) {
  return __builtin_amdgcn_wmma_f32_16x16x32_f16(false, a, false, b, (short)0, c, false, false);
}

// A fragment per CDNA5 16-bit A 16x32 layout:
// lane<16: elems 0..7 = K k0..k0+7, elems 8..15 = K k0+16..k0+23 (row M=lane)
// lane>=16: shifted by 8 in K (sel=8 halves), row M=lane-16
__device__ __forceinline__ v16h afrag(const _Float16* rowbase, int koff, int sel) {
  v8h lo = *(const v8h*)(rowbase + koff + sel);
  v8h hi = *(const v8h*)(rowbase + koff + 16 + sel);
  v16h a;
#pragma unroll
  for (int i = 0; i < 8; ++i) { a[i] = lo[i]; a[i + 8] = hi[i]; }
  return a;
}

// B fragment: weights pre-packed so each lane reads 16 contiguous halves (32B)
__device__ __forceinline__ v16h bfrag(const _Float16* Wp, int kc, int NT, int nt, int lane) {
  return *(const v16h*)(Wp + (size_t)(((kc * NT) + nt) * 32 + lane) * 16);
}

__global__ void zero_f32(float* p, int n) {
  int t = blockIdx.x * blockDim.x + threadIdx.x;
  if (t < n) p[t] = 0.f;
}

// Pack f32 weight matrix (K x Ncols, row-major) into WMMA B-fragment order (f16),
// zero-padding K up to Kpad (multiple of 32).
__global__ void pack_b(const float* __restrict__ W, _Float16* __restrict__ out,
                       int K, int Kpad, int Ncols) {
  int o = blockIdx.x * blockDim.x + threadIdx.x;
  int NT = Ncols >> 4;
  int total = (Kpad >> 5) * NT * 512;
  if (o >= total) return;
  int e    = o & 15;
  int lane = (o >> 4) & 31;
  int rest = o >> 9;
  int nt = rest % NT;
  int kc = rest / NT;
  int n = nt * 16 + (lane & 15);
  int k = kc * 32 + ((e >> 3) << 4) + ((lane >> 4) << 3) + (e & 7);
  out[o] = (k < K) ? (_Float16)W[(size_t)k * Ncols + n] : (_Float16)0.f;
}

__global__ void coord_stats(const float* __restrict__ F, float* __restrict__ stats) {
  int t = blockIdx.x * blockDim.x + threadIdx.x;
  if (t >= Nn) return;
  for (int d = 0; d < 3; ++d) {
    float v = F[(size_t)t * 19 + d];
    atomicAdd(&stats[d], v);
    atomicAdd(&stats[3 + d], v * v);
  }
}

__global__ void finalize_stats(const float* __restrict__ stats, float* __restrict__ ms) {
  if (threadIdx.x == 0 && blockIdx.x == 0) {
    for (int d = 0; d < 3; ++d) {
      float mean = stats[d] / (float)Nn;
      float var  = (stats[3 + d] - (float)Nn * mean * mean) / (float)(Nn - 1); // ddof=1
      ms[d] = mean;
      ms[3 + d] = sqrtf(var > 0.f ? var : 0.f);
    }
  }
}

__global__ void normalize_cz(const float* __restrict__ F, const float* __restrict__ ms,
                             float* __restrict__ cz) {
  int t = blockIdx.x * blockDim.x + threadIdx.x;
  if (t >= Nn * 3) return;
  int n = t / 3, d = t % 3;
  cz[t] = (F[(size_t)n * 19 + d] - ms[d]) / (ms[3 + d] + EPSc);
}

// Point MLP (16->256->64) + input layer (67->128), one wave per node. h stored f16.
__global__ void node_mlp(const float* __restrict__ F, const float* __restrict__ cz,
                         const float* __restrict__ pw1, const float* __restrict__ pb1,
                         const float* __restrict__ pw2, const float* __restrict__ pb2,
                         const float* __restrict__ inw, const float* __restrict__ inb,
                         _Float16* __restrict__ h_out) {
  __shared__ float hid[8][256];
  __shared__ float pbuf[8][64];
  const int wave = threadIdx.x >> 5, lane = threadIdx.x & 31;
  const int node = blockIdx.x * 8 + wave;
  float fk[16];
#pragma unroll
  for (int k = 0; k < 16; ++k) fk[k] = F[(size_t)node * 19 + 3 + k];
#pragma unroll
  for (int s = 0; s < 8; ++s) {
    int o = lane + 32 * s;
    float acc = pb1[o];
    for (int k = 0; k < 16; ++k) acc += fk[k] * pw1[k * 256 + o];
    hid[wave][o] = acc > 0.f ? acc : 0.f;
  }
  __syncthreads();
#pragma unroll
  for (int s = 0; s < 2; ++s) {
    int o = lane + 32 * s;
    float acc = pb2[o];
    for (int k = 0; k < 256; ++k) acc += hid[wave][k] * pw2[k * 64 + o];
    pbuf[wave][o] = acc > 0.f ? acc : 0.f;
  }
  __syncthreads();
  float c3[3];
  for (int d = 0; d < 3; ++d) c3[d] = cz[node * 3 + d];
#pragma unroll
  for (int s = 0; s < 4; ++s) {
    int o = lane + 32 * s;
    float acc = inb[o];
    for (int d = 0; d < 3; ++d) acc += c3[d] * inw[d * Hd + o];
    for (int k = 0; k < 64; ++k) acc += pbuf[wave][k] * inw[(3 + k) * Hd + o];
    h_out[(size_t)node * Hd + o] = (_Float16)(acc > 0.f ? acc : 0.f);
  }
}

__global__ void deg_count(const int* __restrict__ ed, float* __restrict__ deg) {
  int t = blockIdx.x * blockDim.x + threadIdx.x;
  if (t < EDIR) atomicAdd(&deg[ed[EDIR + t]], 1.f);
}

// segment_sum(h[src], dst): 16 lanes per edge, 8 f16 channels each (16B gather),
// f32 atomic accumulation.
__global__ void scatter_h(const _Float16* __restrict__ h, const int* __restrict__ ed,
                          float* __restrict__ agg) {
  int t = blockIdx.x * blockDim.x + threadIdx.x;      // EDIR*16 = 25.6M threads
  int e  = t >> 4;
  int c0 = (t & 15) * 8;
  int s = ed[e], d = ed[EDIR + e];
  v8h v = *(const v8h*)(h + (size_t)s * Hd + c0);
  float* a = agg + (size_t)d * Hd + c0;
#pragma unroll
  for (int q = 0; q < 8; ++q) atomicAdd(a + q, (float)v[q]);
}

// h_new = relu(h @ Ws + bs + (agg/deg) @ Wn + bn); 16 rows/block, 8 waves x 16 cols.
#define APITCH 144
__global__ void gnn_layer(const _Float16* __restrict__ h_in, const float* __restrict__ agg,
                          const float* __restrict__ deg,
                          const _Float16* __restrict__ WpS, const _Float16* __restrict__ WpN,
                          const float* __restrict__ bS, const float* __restrict__ bN,
                          _Float16* __restrict__ h_out) {
  __shared__ __align__(16) _Float16 Ah[16 * APITCH];
  __shared__ __align__(16) _Float16 Ag[16 * APITCH];
  const int t = threadIdx.x;
  const int row0 = blockIdx.x * 16;
  {
    // 256 threads cover 16 rows x 16 chunks of 8 halves exactly
    int m = t >> 4, ch = (t & 15) * 8;
    int r = row0 + m;
    *(v8h*)(Ah + m * APITCH + ch) = *(const v8h*)(h_in + (size_t)r * Hd + ch);
    float dg = deg[r]; dg = dg < 1.f ? 1.f : dg;
    float4 g0 = *(const float4*)(agg + (size_t)r * Hd + ch);
    float4 g1 = *(const float4*)(agg + (size_t)r * Hd + ch + 4);
    v8h gg;
    gg[0] = (_Float16)(g0.x / dg); gg[1] = (_Float16)(g0.y / dg);
    gg[2] = (_Float16)(g0.z / dg); gg[3] = (_Float16)(g0.w / dg);
    gg[4] = (_Float16)(g1.x / dg); gg[5] = (_Float16)(g1.y / dg);
    gg[6] = (_Float16)(g1.z / dg); gg[7] = (_Float16)(g1.w / dg);
    *(v8h*)(Ag + m * APITCH + ch) = gg;
  }
  __syncthreads();
  const int wave = t >> 5, lane = t & 31;
  const int col = wave * 16 + (lane & 15);
  const int sel = (lane >> 4) * 8;
  const _Float16* rbh = Ah + (lane & 15) * APITCH;
  const _Float16* rbg = Ag + (lane & 15) * APITCH;
  const float bias = bS[col] + bN[col];
  v8f acc;
#pragma unroll
  for (int r = 0; r < 8; ++r) acc[r] = bias;
#pragma unroll
  for (int kc = 0; kc < 4; ++kc) {
    acc = wmma16(afrag(rbh, kc * 32, sel), bfrag(WpS, kc, 8, wave, lane), acc);
    acc = wmma16(afrag(rbg, kc * 32, sel), bfrag(WpN, kc, 8, wave, lane), acc);
  }
#pragma unroll
  for (int r = 0; r < 8; ++r) {
    int M = r + sel;                 // C/D layout: VGPR r -> M=r (lanes 0-15) / r+8
    float v = acc[r]; v = v > 0.f ? v : 0.f;
    h_out[(size_t)(row0 + M) * Hd + col] = (_Float16)v;
  }
}

// Fused edge MLP: feats(387, padded 416) ->256 ->256 ->1, 32 edges/block.
// Each wave: 2 M-tiles x 2 N-tiles; B fragments reused across both M-tiles.
#define EM  32
#define FP  432
#define XP  272
__global__ void edge_mlp(const _Float16* __restrict__ h, const float* __restrict__ cz,
                         const int* __restrict__ eu,
                         const _Float16* __restrict__ W1p, const float* __restrict__ b1,
                         const _Float16* __restrict__ W2p, const float* __restrict__ b2,
                         const float* __restrict__ w3, const float* __restrict__ b3,
                         float* __restrict__ out) {
  __shared__ __align__(16) _Float16 feats[EM * FP];  // 27648 B
  __shared__ __align__(16) _Float16 x1[EM * XP];     // 17408 B
  __shared__ __align__(16) _Float16 x2[EM * XP];     // 17408 B
  __shared__ float part[EM * 8];                     //  1024 B  (total 63488 B)
  const int t = threadIdx.x;
  const int e0 = blockIdx.x * EM;
  {
    int m = t >> 3, l = t & 7;   // 8 threads per edge
    int i = eu[e0 + m], j = eu[EUND + e0 + m];
    const v8h* hp = (const v8h*)(h + (size_t)i * Hd);
    const v8h* hq = (const v8h*)(h + (size_t)j * Hd);
#pragma unroll
    for (int ch = 0; ch < 2; ++ch) {
      int c = l + ch * 8;                       // chunk index 0..15
      v8h a = hp[c], b = hq[c], dd;
#pragma unroll
      for (int q = 0; q < 8; ++q) dd[q] = (_Float16)fabsf((float)a[q] - (float)b[q]);
      *(v8h*)(feats + m * FP + c * 8)       = a;
      *(v8h*)(feats + m * FP + 128 + c * 8) = b;
      *(v8h*)(feats + m * FP + 256 + c * 8) = dd;
    }
  }
  if (t < EM) {
    int m = t;
    int i = eu[e0 + m], j = eu[EUND + e0 + m];
    for (int d = 0; d < 32; ++d) {            // coords 384..386, zero pad 387..415
      float v = (d < 3) ? (cz[i * 3 + d] - cz[j * 3 + d]) : 0.f;
      feats[m * FP + 384 + d] = (_Float16)v;
    }
  }
  __syncthreads();
  const int wave = t >> 5, lane = t & 31;
  const int sel = (lane >> 4) * 8;
  const int nt0 = wave * 2, nt1 = wave * 2 + 1;
  const int c0 = nt0 * 16 + (lane & 15), c1 = nt1 * 16 + (lane & 15);
  const _Float16* rb0 = feats + (lane & 15) * FP;
  const _Float16* rb1 = feats + ((lane & 15) + 16) * FP;
  v8f acc00, acc01, acc10, acc11;
#pragma unroll
  for (int r = 0; r < 8; ++r) {
    acc00[r] = b1[c0]; acc01[r] = b1[c1];
    acc10[r] = b1[c0]; acc11[r] = b1[c1];
  }
#pragma unroll
  for (int kc = 0; kc < 13; ++kc) {
    v16h bA = bfrag(W1p, kc, 16, nt0, lane);
    v16h bB = bfrag(W1p, kc, 16, nt1, lane);
    v16h a0 = afrag(rb0, kc * 32, sel);
    v16h a1 = afrag(rb1, kc * 32, sel);
    acc00 = wmma16(a0, bA, acc00);
    acc01 = wmma16(a0, bB, acc01);
    acc10 = wmma16(a1, bA, acc10);
    acc11 = wmma16(a1, bB, acc11);
  }
#pragma unroll
  for (int r = 0; r < 8; ++r) {
    int M0 = r + sel, M1 = r + sel + 16;
    x1[M0 * XP + c0] = (_Float16)(acc00[r] > 0.f ? acc00[r] : 0.f);
    x1[M0 * XP + c1] = (_Float16)(acc01[r] > 0.f ? acc01[r] : 0.f);
    x1[M1 * XP + c0] = (_Float16)(acc10[r] > 0.f ? acc10[r] : 0.f);
    x1[M1 * XP + c1] = (_Float16)(acc11[r] > 0.f ? acc11[r] : 0.f);
  }
  __syncthreads();
  const _Float16* rc0 = x1 + (lane & 15) * XP;
  const _Float16* rc1 = x1 + ((lane & 15) + 16) * XP;
#pragma unroll
  for (int r = 0; r < 8; ++r) {
    acc00[r] = b2[c0]; acc01[r] = b2[c1];
    acc10[r] = b2[c0]; acc11[r] = b2[c1];
  }
#pragma unroll
  for (int kc = 0; kc < 8; ++kc) {
    v16h bA = bfrag(W2p, kc, 16, nt0, lane);
    v16h bB = bfrag(W2p, kc, 16, nt1, lane);
    v16h a0 = afrag(rc0, kc * 32, sel);
    v16h a1 = afrag(rc1, kc * 32, sel);
    acc00 = wmma16(a0, bA, acc00);
    acc01 = wmma16(a0, bB, acc01);
    acc10 = wmma16(a1, bA, acc10);
    acc11 = wmma16(a1, bB, acc11);
  }
#pragma unroll
  for (int r = 0; r < 8; ++r) {
    int M0 = r + sel, M1 = r + sel + 16;
    x2[M0 * XP + c0] = (_Float16)(acc00[r] > 0.f ? acc00[r] : 0.f);
    x2[M0 * XP + c1] = (_Float16)(acc01[r] > 0.f ? acc01[r] : 0.f);
    x2[M1 * XP + c0] = (_Float16)(acc10[r] > 0.f ? acc10[r] : 0.f);
    x2[M1 * XP + c1] = (_Float16)(acc11[r] > 0.f ? acc11[r] : 0.f);
  }
  __syncthreads();
  {
    int m = t >> 3, seg = t & 7;   // 8 partials per edge, 32 channels each
    float s = 0.f;
#pragma unroll
    for (int q = 0; q < 32; ++q) s += (float)x2[m * XP + seg * 32 + q] * w3[seg * 32 + q];
    part[m * 8 + seg] = s;
  }
  __syncthreads();
  if (t < EM) {
    float s = b3[0];
#pragma unroll
    for (int q = 0; q < 8; ++q) s += part[t * 8 + q];
    out[e0 + t] = s;
  }
}

extern "C" void kernel_launch(void* const* d_in, const int* in_sizes, int n_in,
                              void* d_out, int out_size, void* d_ws, size_t ws_size,
                              hipStream_t stream) {
  const float* F   = (const float*)d_in[0];
  const int*   eu  = (const int*)d_in[1];
  const int*   ed  = (const int*)d_in[2];
  const float* pw1 = (const float*)d_in[3];
  const float* pb1 = (const float*)d_in[4];
  const float* pw2 = (const float*)d_in[5];
  const float* pb2 = (const float*)d_in[6];
  const float* inw = (const float*)d_in[7];
  const float* inb = (const float*)d_in[8];
  const float* lws = (const float*)d_in[9];
  const float* lbs = (const float*)d_in[10];
  const float* lwn = (const float*)d_in[11];
  const float* lbn = (const float*)d_in[12];
  const float* ew1 = (const float*)d_in[13];
  const float* eb1 = (const float*)d_in[14];
  const float* ew2 = (const float*)d_in[15];
  const float* eb2 = (const float*)d_in[16];
  const float* ew3 = (const float*)d_in[17];
  const float* eb3 = (const float*)d_in[18];
  float* out = (float*)d_out;
  (void)in_sizes; (void)n_in; (void)out_size;

  char* ws = (char*)d_ws;
  size_t off = 0;
  auto alloc = [&](size_t bytes) -> void* {
    off = (off + 255) & ~(size_t)255;
    void* p = ws + off;
    off += bytes;
    return p;
  };
  float* stats = (float*)alloc(8 * 4);
  float* ms    = (float*)alloc(8 * 4);
  float* cz    = (float*)alloc((size_t)Nn * 3 * 4);
  float* deg   = (float*)alloc((size_t)Nn * 4);
  _Float16* hA = (_Float16*)alloc((size_t)Nn * Hd * 2);
  _Float16* hB = (_Float16*)alloc((size_t)Nn * Hd * 2);
  float* agg   = (float*)alloc((size_t)Nn * Hd * 4);
  _Float16* lwsP = (_Float16*)alloc((size_t)Ld * Hd * Hd * 2);
  _Float16* lwnP = (_Float16*)alloc((size_t)Ld * Hd * Hd * 2);
  _Float16* ew1P = (_Float16*)alloc((size_t)416 * 256 * 2);
  _Float16* ew2P = (_Float16*)alloc((size_t)256 * 256 * 2);
  (void)ws_size; // total ~53 MB

  zero_f32<<<1, 64, 0, stream>>>(stats, 8);
  zero_f32<<<(Nn + 255) / 256, 256, 0, stream>>>(deg, Nn);

  for (int l = 0; l < Ld; ++l) {
    pack_b<<<(Hd * Hd + 255) / 256, 256, 0, stream>>>(lws + (size_t)l * Hd * Hd,
                                                      lwsP + (size_t)l * Hd * Hd, Hd, Hd, Hd);
    pack_b<<<(Hd * Hd + 255) / 256, 256, 0, stream>>>(lwn + (size_t)l * Hd * Hd,
                                                      lwnP + (size_t)l * Hd * Hd, Hd, Hd, Hd);
  }
  pack_b<<<(416 * 256 + 255) / 256, 256, 0, stream>>>(ew1, ew1P, 387, 416, 256);
  pack_b<<<(256 * 256 + 255) / 256, 256, 0, stream>>>(ew2, ew2P, 256, 256, 256);

  coord_stats<<<(Nn + 255) / 256, 256, 0, stream>>>(F, stats);
  finalize_stats<<<1, 1, 0, stream>>>(stats, ms);
  normalize_cz<<<(Nn * 3 + 255) / 256, 256, 0, stream>>>(F, ms, cz);

  node_mlp<<<Nn / 8, 256, 0, stream>>>(F, cz, pw1, pb1, pw2, pb2, inw, inb, hA);
  deg_count<<<(EDIR + 255) / 256, 256, 0, stream>>>(ed, deg);

  _Float16* hc = hA; _Float16* hn = hB;
  for (int l = 0; l < Ld; ++l) {
    zero_f32<<<(Nn * Hd + 255) / 256, 256, 0, stream>>>(agg, Nn * Hd);
    scatter_h<<<(EDIR * 16) / 256, 256, 0, stream>>>(hc, ed, agg);
    gnn_layer<<<Nn / 16, 256, 0, stream>>>(hc, agg, deg,
        lwsP + (size_t)l * Hd * Hd, lwnP + (size_t)l * Hd * Hd,
        lbs + l * Hd, lbn + l * Hd, hn);
    _Float16* tmp = hc; hc = hn; hn = tmp;
  }
  edge_mlp<<<EUND / EM, 256, 0, stream>>>(hc, cz, eu, ew1P, eb1, ew2P, eb2, ew3, eb3, out);
}
